// SiglipAttention_55843164783035
// MI455X (gfx1250) — compile-verified, matrix operand order checked by reading, not run on hardware
//
#include <hip/hip_runtime.h>
#include <hip/hip_bf16.h>

// ---------------- problem constants ----------------
#define B_   2
#define S_   4096
#define D_   1152
#define H_   16
#define HD_  72
#define HDP_ 96            // head dim padded to 3*32 for K32 WMMA steps
#define M_   (B_*S_)       // 8192 rows
#define SCALE_ 0.1178511301977579f  // 72^-0.5

typedef __attribute__((ext_vector_type(16))) __bf16 v16bf;
typedef __attribute__((ext_vector_type(8)))  float  v8f;
typedef __attribute__((ext_vector_type(4)))  unsigned int v4u;
typedef __attribute__((ext_vector_type(8)))  int v8i_;
typedef __attribute__((ext_vector_type(4)))  int v4i_;

union ABFrag { v16bf v; uint4 u[2]; };

static __device__ inline v8f wmma_bf16(v16bf a, v16bf b, v8f c) {
    return __builtin_amdgcn_wmma_f32_16x16x32_bf16(false, a, false, b, (short)0, c, false, false);
}

// Async global->LDS copy of 16 bytes per lane (ASYNCcnt-tracked).
static __device__ inline void async_b128(void* lds_ptr, const void* gptr) {
    unsigned int lds = (unsigned int)(unsigned long long)lds_ptr;
    asm volatile("global_load_async_to_lds_b128 %0, %1, off"
                 :: "v"(lds), "v"(gptr) : "memory");
}

static __device__ inline void wait_async0() {
    asm volatile("s_wait_asynccnt 0x0" ::: "memory");
}

// Tensor Data Mover: 2-D tile DMA global->LDS with hardware LDS padding.
// All dims/strides in 4-byte elements (data_size = 4B). Issued by one wave.
// pad: after every 16 dwords stored, skip 4 dwords of LDS (interval code 3, amount code 3).
static __device__ inline void tdm_load_2d(unsigned int lds_addr, const void* gaddr,
                                          unsigned int tensor_w, unsigned int tensor_h,
                                          unsigned int tile_w, unsigned int tile_h,
                                          unsigned int row_stride, unsigned int pad_en)
{
    unsigned long long ga = (unsigned long long)gaddr;
    v4u g0;
    g0[0] = 1u;                                            // count=1, user mode, no gather
    g0[1] = lds_addr;                                      // LDS byte address
    g0[2] = (unsigned int)(ga & 0xFFFFFFFFu);              // global_addr[31:0]
    g0[3] = (unsigned int)((ga >> 32) & 0x01FFFFFFu)       // global_addr[56:32]
          | 0x80000000u;                                   // type = 2 ("image")

    v8i_ g1;
    g1[0] = (int)((2u << 16)                               // data_size = 4 bytes
                | (pad_en << 20)                           // pad_enable
                | (3u << 22)                               // pad_interval: 16 dwords
                | (3u << 25));                             // pad_amount: 4 dwords
    g1[1] = (int)((tensor_w & 0xFFFFu) << 16);             // tensor_dim0[15:0] @ bits 63:48
    g1[2] = (int)(((tensor_w >> 16) & 0xFFFFu)             // tensor_dim0[31:16]
                | ((tensor_h & 0xFFFFu) << 16));           // tensor_dim1[15:0]
    g1[3] = (int)(((tensor_h >> 16) & 0xFFFFu)             // tensor_dim1[31:16]
                | ((tile_w & 0xFFFFu) << 16));             // tile_dim0
    g1[4] = (int)(tile_h & 0xFFFFu);                       // tile_dim1 (tile_dim2 = 0)
    g1[5] = (int)row_stride;                               // tensor_dim0_stride[31:0]
    g1[6] = 0;                                             // stride hi / dim1_stride lo
    g1[7] = 0;

    v4i_ z4; z4[0] = 0; z4[1] = 0; z4[2] = 0; z4[3] = 0;   // 2-D: groups 2/3 unused
    v8i_ z8;
#pragma unroll
    for (int i = 0; i < 8; i++) z8[i] = 0;
    __builtin_amdgcn_tensor_load_to_lds(g0, g1, z4, z4, z8, 0);
}

static __device__ inline void wait_tensor0() {
    __builtin_amdgcn_s_wait_tensorcnt(0);
}

// ---------------- fp32 -> bf16 convert ----------------
__global__ void k_cvt_bf16(const float* __restrict__ src, __bf16* __restrict__ dst, int n) {
    int i = blockIdx.x * blockDim.x + threadIdx.x;
    int stride = gridDim.x * blockDim.x;
    for (; i < n; i += stride) dst[i] = (__bf16)src[i];
}

// ---------------- zero fill (uint4 granularity) ----------------
__global__ void k_zero16(uint4* __restrict__ dst, int n4) {
    int i = blockIdx.x * blockDim.x + threadIdx.x;
    int stride = gridDim.x * blockDim.x;
    uint4 z; z.x = 0; z.y = 0; z.z = 0; z.w = 0;
    for (; i < n4; i += stride) dst[i] = z;
}

// ---------------- tiled bf16 GEMM: Y[M,N] = A[M,K] * W[N,K]^T + bias ----------------
// mode 0: write bf16 to per-head layout  [(b*H+h)*S + s]*HDP + hd      (Q, K)
// mode 1: write bf16 to transposed head  [(b*H+h)*HDP + hd]*S + s      (V)
// mode 2: write fp32 to [M, D] (+bias)                                  (final proj)
__global__ __launch_bounds__(128)
void k_gemm(const __bf16* __restrict__ A, const __bf16* __restrict__ W,
            const float* __restrict__ bias, void* __restrict__ out, int mode)
{
    // TDM pads every 16 dwords with 4 dwords -> row stride 40 bf16
    __shared__ __align__(16) __bf16 As[2][64][40];
    __shared__ __align__(16) __bf16 Ws[2][64][40];

    const int t    = threadIdx.x;
    const int wave = t >> 5;
    const int lane = t & 31;
    const int l15  = lane & 15;
    const int hi   = lane >> 4;
    const int kb   = hi * 8;

    const int m0 = blockIdx.y * 64;
    const int n0 = blockIdx.x * 64;
    const int qr = (wave >> 1) * 32;   // wave quadrant within 64x64 tile
    const int qc = (wave & 1) * 32;

    v8f acc[2][2];
#pragma unroll
    for (int i = 0; i < 2; i++)
#pragma unroll
        for (int j = 0; j < 2; j++) acc[i][j] = v8f{0,0,0,0,0,0,0,0};

    // TDM-stage one 64x32 A tile + 64x32 W tile into buffer p (wave 0 only)
    auto stage_tdm = [&](int p, int k0) {
        tdm_load_2d((unsigned int)(unsigned long long)&As[p][0][0],
                    A + (size_t)m0 * D_ + k0,
                    /*tensor_w*/ D_ / 2, /*tensor_h*/ 64,
                    /*tile_w*/ 16, /*tile_h*/ 64,
                    /*row_stride*/ D_ / 2, /*pad*/ 1);
        tdm_load_2d((unsigned int)(unsigned long long)&Ws[p][0][0],
                    W + (size_t)n0 * D_ + k0,
                    D_ / 2, 64, 16, 64, D_ / 2, 1);
    };

    if (wave == 0) stage_tdm(0, 0);

    for (int k0 = 0; k0 < D_; k0 += 32) {
        const int p = (k0 >> 5) & 1;
        if (wave == 0) wait_tensor0();     // tile-k0 DMA complete
        __syncthreads();                   // publish to all waves
        if (wave == 0 && k0 + 32 < D_) stage_tdm(p ^ 1, k0 + 32);

        ABFrag a[2], b[2];
#pragma unroll
        for (int i = 0; i < 2; i++) {
            const int m = qr + i * 16 + l15;
            a[i].u[0] = *(const uint4*)&As[p][m][kb];
            a[i].u[1] = *(const uint4*)&As[p][m][16 + kb];
        }
#pragma unroll
        for (int j = 0; j < 2; j++) {
            const int n = qc + j * 16 + l15;
            b[j].u[0] = *(const uint4*)&Ws[p][n][kb];
            b[j].u[1] = *(const uint4*)&Ws[p][n][16 + kb];
        }
#pragma unroll
        for (int i = 0; i < 2; i++)
#pragma unroll
            for (int j = 0; j < 2; j++)
                acc[i][j] = wmma_bf16(a[i].v, b[j].v, acc[i][j]);
    }

    // epilogue
#pragma unroll
    for (int i = 0; i < 2; i++) {
#pragma unroll
        for (int j = 0; j < 2; j++) {
#pragma unroll
            for (int r = 0; r < 8; r++) {
                const int mg = m0 + qr + i * 16 + r + hi * 8;
                const int ng = n0 + qc + j * 16 + l15;
                const float v = acc[i][j][r] + bias[ng];
                if (mode == 2) {
                    ((float*)out)[(size_t)mg * D_ + ng] = v;
                } else {
                    const int bb = mg / S_, s = mg % S_;
                    const int h  = ng / HD_, hd = ng % HD_;
                    __bf16* o = (__bf16*)out;
                    if (mode == 0)
                        o[((size_t)(bb * H_ + h) * S_ + s) * HDP_ + hd] = (__bf16)v;
                    else
                        o[((size_t)(bb * H_ + h) * HDP_ + hd) * S_ + s] = (__bf16)v;
                }
            }
        }
    }
}

// ---------------- flash attention ----------------
// Q, K: bf16 [(b*H+h)*S + s][HDP]; Vt: bf16 [(b*H+h)*HDP + d][S]
// AO: bf16 [b*S + s][D] (head-interleaved, ready for output projection)
__global__ __launch_bounds__(128)
void k_flash(const __bf16* __restrict__ Q, const __bf16* __restrict__ K,
             const __bf16* __restrict__ Vt, __bf16* __restrict__ AO)
{
    __shared__ __align__(16) __bf16 Qs[64][104];       // async-staged, stride 104
    __shared__ __align__(16) __bf16 Ks[2][32][120];    // TDM: 48dw rows -> 60dw stride
    __shared__ __align__(16) __bf16 Vts[2][96][40];    // TDM: 16dw rows -> 20dw stride
    __shared__ __align__(16) __bf16 Ps[4][16][40];     // wave-private P staging

    const int t    = threadIdx.x;
    const int wave = t >> 5;
    const int lane = t & 31;
    const int l15  = lane & 15;
    const int hi   = lane >> 4;
    const int kb   = hi * 8;

    const int bh = blockIdx.y;          // b*H + h
    const int q0 = blockIdx.x * 64;

    // stage Q tile 64x96 via async-to-LDS (all lanes)
    {
        const int row = t >> 1, c0 = (t & 1) * 48;
        const __bf16* g = Q + ((size_t)bh * S_ + q0 + row) * HDP_ + c0;
#pragma unroll
        for (int i = 0; i < 6; i++)
            async_b128(&Qs[row][c0 + i * 8], g + i * 8);
    }

    // TDM-stage K tile 32x96 and Vt tile 96x32 into buffer p (wave 0 only)
    auto stage_kv = [&](int p, int kt) {
        tdm_load_2d((unsigned int)(unsigned long long)&Ks[p][0][0],
                    K + ((size_t)bh * S_ + kt) * HDP_,
                    /*tensor_w*/ HDP_ / 2, /*tensor_h*/ 32,
                    /*tile_w*/ HDP_ / 2, /*tile_h*/ 32,
                    /*row_stride*/ HDP_ / 2, /*pad*/ 1);
        tdm_load_2d((unsigned int)(unsigned long long)&Vts[p][0][0],
                    Vt + (size_t)bh * HDP_ * S_ + kt,
                    /*tensor_w*/ S_ / 2, /*tensor_h*/ HDP_,
                    /*tile_w*/ 16, /*tile_h*/ HDP_,
                    /*row_stride*/ S_ / 2, /*pad*/ 1);
    };

    float mrow[8], lsum[8];
    v8f accO[6];
#pragma unroll
    for (int r = 0; r < 8; r++) { mrow[r] = -1e30f; lsum[r] = 0.f; }
#pragma unroll
    for (int tt = 0; tt < 6; tt++) accO[tt] = v8f{0,0,0,0,0,0,0,0};

    if (wave == 0) stage_kv(0, 0);
    wait_async0();                      // our Q async copies are done

    for (int kt = 0; kt < S_; kt += 32) {
        const int p = (kt >> 5) & 1;
        if (wave == 0) wait_tensor0();  // tile-kt DMA complete
        __syncthreads();                // publish Q (first iter) + K/V tile
        if (wave == 0 && kt + 32 < S_) stage_kv(p ^ 1, kt + 32);

        // scores: 16 query rows x 32 keys, K-dim = 96 (3 WMMA steps), f32 accum
        v8f sc[2];
        sc[0] = v8f{0,0,0,0,0,0,0,0};
        sc[1] = v8f{0,0,0,0,0,0,0,0};
#pragma unroll
        for (int ks = 0; ks < 3; ks++) {
            ABFrag a, b0, b1;
            const int m = wave * 16 + l15;
            a.u[0]  = *(const uint4*)&Qs[m][ks * 32 + kb];
            a.u[1]  = *(const uint4*)&Qs[m][ks * 32 + 16 + kb];
            b0.u[0] = *(const uint4*)&Ks[p][l15][ks * 40 + kb];
            b0.u[1] = *(const uint4*)&Ks[p][l15][ks * 40 + 16 + kb];
            b1.u[0] = *(const uint4*)&Ks[p][16 + l15][ks * 40 + kb];
            b1.u[1] = *(const uint4*)&Ks[p][16 + l15][ks * 40 + 16 + kb];
            sc[0] = wmma_bf16(a.v, b0.v, sc[0]);
            sc[1] = wmma_bf16(a.v, b1.v, sc[1]);
        }

        // online softmax: row r lives at lanes (hi*16..hi*16+15), vgpr r
#pragma unroll
        for (int r = 0; r < 8; r++) {
            const float s0 = sc[0][r] * SCALE_;
            const float s1 = sc[1][r] * SCALE_;
            float mx = fmaxf(s0, s1);
#pragma unroll
            for (int msk = 1; msk < 16; msk <<= 1) mx = fmaxf(mx, __shfl_xor(mx, msk));
            const float newm = fmaxf(mrow[r], mx);
            const float corr = __expf(mrow[r] - newm);
            const float p0 = __expf(s0 - newm);
            const float p1 = __expf(s1 - newm);
            float rs = p0 + p1;
#pragma unroll
            for (int msk = 1; msk < 16; msk <<= 1) rs += __shfl_xor(rs, msk);
            lsum[r] = lsum[r] * corr + rs;
            mrow[r] = newm;
#pragma unroll
            for (int tt = 0; tt < 6; tt++) accO[tt][r] *= corr;
            const int m = r + 8 * hi;
            Ps[wave][m][l15]      = (__bf16)p0;   // wave-private: LDS in-order per wave
            Ps[wave][m][16 + l15] = (__bf16)p1;
        }

        // P (16x32) @ V (32x96): re-layout P via LDS into an A fragment
        ABFrag pa;
        pa.u[0] = *(const uint4*)&Ps[wave][l15][kb];
        pa.u[1] = *(const uint4*)&Ps[wave][l15][16 + kb];
#pragma unroll
        for (int tt = 0; tt < 6; tt++) {
            ABFrag bv;
            const int d = tt * 16 + l15;
            bv.u[0] = *(const uint4*)&Vts[p][d][kb];
            bv.u[1] = *(const uint4*)&Vts[p][d][16 + kb];
            accO[tt] = wmma_bf16(pa.v, bv.v, accO[tt]);
        }
    }

    // epilogue: O / l, drop pad cols, write head-interleaved bf16
    const int bb = bh >> 4, h = bh & 15;
#pragma unroll
    for (int tt = 0; tt < 6; tt++) {
#pragma unroll
        for (int r = 0; r < 8; r++) {
            const int col = tt * 16 + l15;
            if (col < HD_) {
                const int s = q0 + wave * 16 + r + 8 * hi;
                const float inv = 1.f / lsum[r];
                AO[((size_t)bb * S_ + s) * D_ + h * HD_ + col] = (__bf16)(accO[tt][r] * inv);
            }
        }
    }
}

// ---------------- launcher ----------------
extern "C" void kernel_launch(void* const* d_in, const int* in_sizes, int n_in,
                              void* d_out, int out_size, void* d_ws, size_t ws_size,
                              hipStream_t stream)
{
    const float* x   = (const float*)d_in[0];
    const float* q_w = (const float*)d_in[1];
    const float* q_b = (const float*)d_in[2];
    const float* k_w = (const float*)d_in[3];
    const float* k_b = (const float*)d_in[4];
    const float* v_w = (const float*)d_in[5];
    const float* v_b = (const float*)d_in[6];
    const float* o_w = (const float*)d_in[7];
    const float* o_b = (const float*)d_in[8];

    char* ws = (char*)d_ws;
    size_t off = 0;
    auto alloc = [&](size_t bytes) -> void* {
        void* p = ws + off;
        off += (bytes + 255) & ~(size_t)255;
        return p;
    };

    const size_t nX  = (size_t)M_ * D_;              // 9,437,184
    const size_t nW  = (size_t)D_ * D_;              // 1,327,104
    const size_t nHP = (size_t)B_ * H_ * S_ * HDP_;  // 12,582,912

    __bf16* xb  = (__bf16*)alloc(nX * 2);
    __bf16* wqb = (__bf16*)alloc(nW * 2);
    __bf16* wkb = (__bf16*)alloc(nW * 2);
    __bf16* wvb = (__bf16*)alloc(nW * 2);
    __bf16* wob = (__bf16*)alloc(nW * 2);
    __bf16* qp  = (__bf16*)alloc(nHP * 2);
    __bf16* kp  = (__bf16*)alloc(nHP * 2);
    __bf16* vt  = (__bf16*)alloc(nHP * 2);
    __bf16* ao  = (__bf16*)alloc(nX * 2);

    // 1) convert inputs to bf16
    k_cvt_bf16<<<4096, 256, 0, stream>>>(x,   xb,  (int)nX);
    k_cvt_bf16<<<1296, 256, 0, stream>>>(q_w, wqb, (int)nW);
    k_cvt_bf16<<<1296, 256, 0, stream>>>(k_w, wkb, (int)nW);
    k_cvt_bf16<<<1296, 256, 0, stream>>>(v_w, wvb, (int)nW);
    k_cvt_bf16<<<1296, 256, 0, stream>>>(o_w, wob, (int)nW);

    // 2) zero padded staging buffers (head-dim pad 72..95 must be 0)
    const int n4 = (int)(nHP * 2 / 16);
    k_zero16<<<1536, 256, 0, stream>>>((uint4*)qp, n4);
    k_zero16<<<1536, 256, 0, stream>>>((uint4*)kp, n4);
    k_zero16<<<1536, 256, 0, stream>>>((uint4*)vt, n4);

    // 3) projections (bf16 WMMA, f32 accumulate)
    dim3 ggrid(D_ / 64, M_ / 64);
    k_gemm<<<ggrid, 128, 0, stream>>>(xb, wqb, q_b, qp, 0);
    k_gemm<<<ggrid, 128, 0, stream>>>(xb, wkb, k_b, kp, 0);
    k_gemm<<<ggrid, 128, 0, stream>>>(xb, wvb, v_b, vt, 1);

    // 4) fused flash attention
    dim3 fgrid(S_ / 64, B_ * H_);
    k_flash<<<fgrid, 128, 0, stream>>>(qp, kp, vt, ao);

    // 5) output projection -> fp32 d_out
    k_gemm<<<ggrid, 128, 0, stream>>>(ao, wob, o_b, d_out, 2);
}
